// BatchTopKSAE_48644799594818
// MI455X (gfx1250) — compile-verified, matrix operand order checked by reading, not run on hardware
//
#include <hip/hip_runtime.h>
#include <hip/hip_bf16.h>

// ---------------- problem sizes ----------------
#define B_SZ   4096
#define DIN    1024
#define DSAE   16384
#define NTOT   ((size_t)B_SZ * (size_t)DSAE)       // 67,108,864
#define XHAT_ELEMS ((size_t)B_SZ * (size_t)DIN)    // 4,194,304
#define META_OFF (XHAT_ELEMS + NTOT)               // 71,303,168

// ---------------- workspace layout (bytes) ----------------
#define WS_CHIST 0        // 2048 u32 coarse histogram
#define WS_FHIST 8192     // 2048 u32 fine histogram
#define WS_CBIN  16384    // u32 coarse bin
#define WS_REM   16388    // u32 remaining count in coarse bin
#define WS_THR   16392    // f32 threshold
#define WS_PNNZ  16400    // 2048 u32 per-block nnz partials
#define WS_PSUM  24592    // 2048 f32 per-block sum partials

typedef __attribute__((ext_vector_type(16))) __bf16 v16bf;
typedef __attribute__((ext_vector_type(8)))  float  v8f;

union Frag { unsigned u[8]; v16bf v; };

__device__ __forceinline__ unsigned short f2bf(float f) {
  // round-to-nearest-even fp32 -> bf16
  unsigned u = __float_as_uint(f);
  unsigned r = u + 0x7FFFu + ((u >> 16) & 1u);
  return (unsigned short)(r >> 16);
}
__device__ __forceinline__ unsigned pk_bf16(float lo, float hi) {
  return (unsigned)f2bf(lo) | ((unsigned)f2bf(hi) << 16);
}

// ---------------- ws zero ----------------
__global__ void sae_zero_ws(unsigned* __restrict__ ws) {
  int i = blockIdx.x * 256 + threadIdx.x;
  if (i < 4100) ws[i] = 0u;   // chist + fhist + cbin + rem + thr + pad
}

// ---------------- encode: a = relu(x @ W_enc + b_enc), fused coarse hist ----
// Block tile 128x256, 8 waves, each wave 64x64 (4x4 WMMA tiles):
// 16 v_wmma per 8 fragment loads per K-step -> 2x the matrix work per ds byte.
#define BM 128
#define BN 256
#define BK 32
#define ASTR 36   // 72-byte rows: 8B-aligned b64 stores, 18-word bank stride (conflict-free)

__global__ void __launch_bounds__(256)
sae_encode_wmma(const float* __restrict__ x,
                const float* __restrict__ W_enc,
                const float* __restrict__ b_enc,
                float* __restrict__ a_out,          // z region of d_out
                unsigned* __restrict__ chist)
{
  __shared__ unsigned short As[BM * ASTR];          // x tile    [m][k] bf16
  __shared__ unsigned short Bs[BN * ASTR];          // Wenc tile [n][k] bf16 (transposed)
  __shared__ unsigned hist[2048];

  const int t = threadIdx.x;
  for (int i = t; i < 2048; i += 256) hist[i] = 0u;

  const int row0 = blockIdx.y * BM;
  const int col0 = blockIdx.x * BN;

  const int w     = t >> 5;          // wave 0..7
  const int lane  = t & 31;
  const int lhalf = lane >> 4;       // 0: lanes 0-15, 1: lanes 16-31
  const int l16   = lane & 15;
  const int wm    = (w >> 2) * 64;   // wave row offset: 0 / 64
  const int wn    = (w & 3) * 64;    // wave col offset: 0/64/128/192

  const v8f vzero = {0.f,0.f,0.f,0.f,0.f,0.f,0.f,0.f};
  v8f acc[4][4];
  for (int i = 0; i < 4; ++i)
    for (int j = 0; j < 4; ++j) acc[i][j] = vzero;

  for (int kt = 0; kt < DIN; kt += BK) {
    __syncthreads();
    // stage A: 128x32 fp32 -> bf16; float4 loads, packed uint2 (b64) stores
    #pragma unroll
    for (int i = t; i < (BM * BK) / 4; i += 256) {       // 4 iters
      int idx = i * 4;
      int r = idx >> 5, k = idx & 31;                    // k multiple of 4
      const float4 f = *reinterpret_cast<const float4*>(
          &x[(size_t)(row0 + r) * DIN + kt + k]);
      uint2 p; p.x = pk_bf16(f.x, f.y); p.y = pk_bf16(f.z, f.w);
      *reinterpret_cast<uint2*>(&As[r * ASTR + k]) = p;  // 72r+2k = 0 mod 8
    }
    // stage B transposed: Bs[n][k]; float4 global loads along n
    #pragma unroll
    for (int i = t; i < (BK * BN) / 4; i += 256) {       // 8 iters
      int idx = i * 4;
      int k = idx >> 8, n = idx & 255;                   // n multiple of 4
      const float4 f = *reinterpret_cast<const float4*>(
          &W_enc[(size_t)(kt + k) * DSAE + col0 + n]);
      Bs[(n + 0) * ASTR + k] = f2bf(f.x);
      Bs[(n + 1) * ASTR + k] = f2bf(f.y);
      Bs[(n + 2) * ASTR + k] = f2bf(f.z);
      Bs[(n + 3) * ASTR + k] = f2bf(f.w);
    }
    __syncthreads();

    if (kt + BK < DIN) {  // global_prefetch_b8 for the next K tile
      __builtin_prefetch(&x[(size_t)(row0 + (t >> 1)) * DIN + kt + BK], 0, 0);
      __builtin_prefetch(&W_enc[(size_t)(kt + BK + (t >> 6)) * DSAE + col0 + ((t & 63) << 2)], 0, 0);
    }

    // Build fragments per CDNA5 16-bit A/B layout (ISA 7.12.2):
    // lanes<16 hold K {2v,2v+1 | v<4} and {16+2(v-4),..}; lanes>=16 are +8.
    Frag afr[4], bfr[4];
    #pragma unroll
    for (int i = 0; i < 4; ++i) {
      const unsigned short* rp = &As[(wm + 16 * i + l16) * ASTR];
      #pragma unroll
      for (int v = 0; v < 8; ++v) {
        int kk = ((v < 4) ? (2 * v) : (16 + 2 * (v - 4))) + (lhalf ? 8 : 0);
        afr[i].u[v] = *reinterpret_cast<const unsigned*>(rp + kk);
      }
    }
    #pragma unroll
    for (int j = 0; j < 4; ++j) {
      const unsigned short* rp = &Bs[(wn + 16 * j + l16) * ASTR];
      #pragma unroll
      for (int v = 0; v < 8; ++v) {
        int kk = ((v < 4) ? (2 * v) : (16 + 2 * (v - 4))) + (lhalf ? 8 : 0);
        bfr[j].u[v] = *reinterpret_cast<const unsigned*>(rp + kk);
      }
    }
    #pragma unroll
    for (int i = 0; i < 4; ++i)
      #pragma unroll
      for (int j = 0; j < 4; ++j)
        acc[i][j] = __builtin_amdgcn_wmma_f32_16x16x32_bf16(
            false, afr[i].v, false, bfr[j].v, (short)0, acc[i][j], false, false);
  }

  // epilogue: bias + relu + store + coarse histogram (positive-float bits>>20)
  #pragma unroll
  for (int i = 0; i < 4; ++i) {
    #pragma unroll
    for (int j = 0; j < 4; ++j) {
      const int n = col0 + wn + 16 * j + l16;
      const float bias = b_enc[n];
      #pragma unroll
      for (int r = 0; r < 8; ++r) {
        const int m = row0 + wm + 16 * i + 8 * lhalf + r;
        float v = acc[i][j][r] + bias;
        v = v > 0.f ? v : 0.f;
        a_out[(size_t)m * DSAE + n] = v;
        if (v > 0.f) atomicAdd(&hist[__float_as_uint(v) >> 20], 1u);
      }
    }
  }
  __syncthreads();
  for (int i = t; i < 2048; i += 256) {
    unsigned c = hist[i];
    if (c) atomicAdd(&chist[i], c);
  }
}

// ---------------- coarse select ----------------
__global__ void sae_select_coarse(const unsigned* __restrict__ chist,
                                  const int* __restrict__ kptr,
                                  unsigned* __restrict__ cbin,
                                  unsigned* __restrict__ rem)
{
  if (threadIdx.x != 0 || blockIdx.x != 0) return;
  const unsigned k = (unsigned)(*kptr);
  unsigned cum = 0;
  int b;
  for (b = 2047; b >= 1; --b) {
    unsigned c = chist[b];
    if (cum + c >= k) break;
    cum += c;
  }
  *cbin = (unsigned)(b < 1 ? 0 : b); // b==0 => fewer positives than k: accept all
  *rem  = k - cum;
}

// ---------------- fine histogram within coarse bin (float4 streaming) --------
__global__ void __launch_bounds__(256)
sae_fine_hist(const float* __restrict__ a, const unsigned* __restrict__ cbin,
              unsigned* __restrict__ fhist, size_t n4)
{
  __shared__ unsigned hist[2048];
  for (int i = threadIdx.x; i < 2048; i += 256) hist[i] = 0u;
  __syncthreads();
  const unsigned cb = *cbin;
  const float4* __restrict__ a4 = reinterpret_cast<const float4*>(a);
  const size_t stride = (size_t)gridDim.x * blockDim.x;
  for (size_t i = (size_t)blockIdx.x * blockDim.x + threadIdx.x; i < n4; i += stride) {
    const float4 f = a4[i];
    const float vv[4] = {f.x, f.y, f.z, f.w};
    #pragma unroll
    for (int c = 0; c < 4; ++c) {
      float v = vv[c];
      if (v > 0.f) {
        unsigned bits = __float_as_uint(v);
        if ((bits >> 20) == cb) atomicAdd(&hist[(bits >> 9) & 2047u], 1u);
      }
    }
  }
  __syncthreads();
  for (int i = threadIdx.x; i < 2048; i += 256) {
    unsigned c = hist[i];
    if (c) atomicAdd(&fhist[i], c);
  }
}

// ---------------- fine select -> threshold ----------------
__global__ void sae_select_fine(const unsigned* __restrict__ fhist,
                                const unsigned* __restrict__ cbin,
                                const unsigned* __restrict__ rem,
                                float* __restrict__ thr)
{
  if (threadIdx.x != 0 || blockIdx.x != 0) return;
  const unsigned r = *rem;
  unsigned cum = 0;
  int f;
  for (f = 2047; f >= 1; --f) {
    unsigned c = fhist[f];
    if (cum + c >= r) break;
    cum += c;
  }
  if (f < 0) f = 0;
  unsigned bits = ((*cbin) << 20) | ((unsigned)f << 9);
  *thr = __uint_as_float(bits);
}

// ---------------- threshold apply (in place, b128) + partial nnz/sum ---------
__global__ void __launch_bounds__(256)
sae_threshold(float* __restrict__ z, const float* __restrict__ thr,
              unsigned* __restrict__ pnnz, float* __restrict__ psum, size_t n4)
{
  __shared__ unsigned snnz[256];
  __shared__ float    ssum[256];
  const float T = *thr;
  float4* __restrict__ z4 = reinterpret_cast<float4*>(z);
  unsigned cnt = 0; float sum = 0.f;
  const size_t stride = (size_t)gridDim.x * blockDim.x;
  for (size_t i = (size_t)blockIdx.x * blockDim.x + threadIdx.x; i < n4; i += stride) {
    float4 f = z4[i];
    float* vv = reinterpret_cast<float*>(&f);
    #pragma unroll
    for (int c = 0; c < 4; ++c) {
      float v = vv[c];
      bool keep = (v > 0.f) && (v >= T);
      float o = keep ? v : 0.f;
      vv[c] = o;
      cnt += keep ? 1u : 0u;
      sum += o;
    }
    z4[i] = f;
  }
  snnz[threadIdx.x] = cnt; ssum[threadIdx.x] = sum;
  __syncthreads();
  for (int s = 128; s > 0; s >>= 1) {
    if ((int)threadIdx.x < s) {
      snnz[threadIdx.x] += snnz[threadIdx.x + s];
      ssum[threadIdx.x] += ssum[threadIdx.x + s];
    }
    __syncthreads();
  }
  if (threadIdx.x == 0) { pnnz[blockIdx.x] = snnz[0]; psum[blockIdx.x] = ssum[0]; }
}

// ---------------- sparse decode: x_hat = z @ W_dec + b_dec ----------------
// One workgroup per row; deterministic ballot/popc compaction into an LDS queue
// (wave-segment ordered), then all threads walk the queue in fixed order.
__global__ void __launch_bounds__(256)
sae_decode(const float* __restrict__ z, const float* __restrict__ W_dec,
           const float* __restrict__ b_dec, float* __restrict__ xhat)
{
  __shared__ unsigned qidx[2048];
  __shared__ float    qval[2048];
  __shared__ unsigned wcnt[8];
  __shared__ unsigned woff[9];

  const int t = threadIdx.x;
  const int w = t >> 5, lane = t & 31;
  const int row = blockIdx.x;
  const float* zrow = z + (size_t)row * DSAE;

  float a0 = 0.f, a1 = 0.f, a2 = 0.f, a3 = 0.f;

  for (int c0 = 0; c0 < DSAE; c0 += 2048) {
    // pass 1: wave covers 256 contiguous elements via two b128 loads per lane
    const int base = c0 + w * 256;
    const float4 f0 = *reinterpret_cast<const float4*>(zrow + base + 4 * lane);
    const float4 f1 = *reinterpret_cast<const float4*>(zrow + base + 128 + 4 * lane);
    float vals[8] = {f0.x, f0.y, f0.z, f0.w, f1.x, f1.y, f1.z, f1.w};
    unsigned cnt = 0;
    #pragma unroll
    for (int s = 0; s < 8; ++s) {
      unsigned long long m = __ballot(vals[s] > 0.f);
      cnt += (unsigned)__popcll(m);
    }
    if (lane == 0) wcnt[w] = cnt;
    __syncthreads();
    if (t == 0) {
      unsigned o = 0;
      for (int i = 0; i < 8; ++i) { woff[i] = o; o += wcnt[i]; }
      woff[8] = o;
    }
    __syncthreads();
    // pass 2: deterministic compacted scatter at exact prefix positions
    unsigned off = woff[w];
    #pragma unroll
    for (int s = 0; s < 8; ++s) {
      const float v = vals[s];
      unsigned long long m  = __ballot(v > 0.f);
      unsigned long long lt = m & ((1ull << lane) - 1ull);
      if (v > 0.f) {
        unsigned p = off + (unsigned)__popcll(lt);
        qidx[p] = (unsigned)(base + (s >> 2) * 128 + 4 * lane + (s & 3));
        qval[p] = v;
      }
      off += (unsigned)__popcll(m);
    }
    __syncthreads();
    // accumulate: thread t owns output columns 4t..4t+3; W_dec row reads are
    // perfectly coalesced float4 (4 KB/row), L2-resident (W_dec = 64MB < 192MB L2)
    const unsigned total = woff[8];
    for (unsigned q = 0; q < total; ++q) {
      const unsigned idx = qidx[q];
      const float v = qval[q];
      const float4 wv = *reinterpret_cast<const float4*>(W_dec + (size_t)idx * DIN + 4 * t);
      a0 += v * wv.x; a1 += v * wv.y; a2 += v * wv.z; a3 += v * wv.w;
    }
    __syncthreads();
  }
  const float4 bd = *reinterpret_cast<const float4*>(b_dec + 4 * t);
  float4 o; o.x = a0 + bd.x; o.y = a1 + bd.y; o.z = a2 + bd.z; o.w = a3 + bd.w;
  *reinterpret_cast<float4*>(xhat + (size_t)row * DIN + 4 * t) = o;
}

// ---------------- metrics ----------------
__global__ void __launch_bounds__(256)
sae_metrics(const unsigned* __restrict__ pnnz, const float* __restrict__ psum,
            float* __restrict__ meta)
{
  __shared__ unsigned sn[256];
  __shared__ float    ss[256];
  const int t = threadIdx.x;
  unsigned cnt = 0; float sum = 0.f;
  for (int i = 0; i < 8; ++i) { cnt += pnnz[t * 8 + i]; sum += psum[t * 8 + i]; }
  sn[t] = cnt; ss[t] = sum;
  __syncthreads();
  for (int s = 128; s > 0; s >>= 1) {
    if (t < s) { sn[t] += sn[t + s]; ss[t] += ss[t + s]; }
    __syncthreads();
  }
  if (t == 0) {
    const unsigned nnz = sn[0];
    meta[0] = (float)nnz / (float)NTOT;                 // frac_nnz
    meta[1] = ss[0] / fmaxf((float)nnz, 1.0f);          // mean_active
    meta[2] = (float)nnz;                               // nnz
  }
}

// ---------------- launcher ----------------
extern "C" void kernel_launch(void* const* d_in, const int* in_sizes, int n_in,
                              void* d_out, int out_size, void* d_ws, size_t ws_size,
                              hipStream_t stream)
{
  const float* x     = (const float*)d_in[0];
  const float* W_enc = (const float*)d_in[1];
  const float* b_enc = (const float*)d_in[2];
  const float* W_dec = (const float*)d_in[3];
  const float* b_dec = (const float*)d_in[4];
  const int*   kptr  = (const int*)d_in[5];

  float* out  = (float*)d_out;
  float* xhat = out;
  float* zbuf = out + XHAT_ELEMS;
  float* meta = out + META_OFF;

  unsigned char* ws = (unsigned char*)d_ws;
  unsigned* chist = (unsigned*)(ws + WS_CHIST);
  unsigned* fhist = (unsigned*)(ws + WS_FHIST);
  unsigned* cbin  = (unsigned*)(ws + WS_CBIN);
  unsigned* rem   = (unsigned*)(ws + WS_REM);
  float*    thr   = (float*)   (ws + WS_THR);
  unsigned* pnnz  = (unsigned*)(ws + WS_PNNZ);
  float*    psum  = (float*)   (ws + WS_PSUM);

  sae_zero_ws<<<17, 256, 0, stream>>>((unsigned*)ws);

  dim3 gA(DSAE / BN, B_SZ / BM);   // 64 x 32 workgroups, 128x256 tiles
  sae_encode_wmma<<<gA, 256, 0, stream>>>(x, W_enc, b_enc, zbuf, chist);

  sae_select_coarse<<<1, 32, 0, stream>>>(chist, kptr, cbin, rem);
  sae_fine_hist<<<2048, 256, 0, stream>>>(zbuf, cbin, fhist, NTOT / 4);
  sae_select_fine<<<1, 32, 0, stream>>>(fhist, cbin, rem, thr);
  sae_threshold<<<2048, 256, 0, stream>>>(zbuf, thr, pnnz, psum, NTOT / 4);
  sae_decode<<<B_SZ, 256, 0, stream>>>(zbuf, W_dec, b_dec, xhat);
  sae_metrics<<<1, 256, 0, stream>>>(pnnz, psum, meta);
}
